// MP_Attention_26654567039403
// MI455X (gfx1250) — matrix operand order repaired and checked by reading, not compiled, not run on hardware
//
#include <hip/hip_runtime.h>
#include <hip/hip_bf16.h>

typedef __bf16 bf16_t;
typedef __attribute__((ext_vector_type(16))) __bf16 v16bf;
typedef __attribute__((ext_vector_type(8)))  float  v8f;

#define MP_EPS 1e-4f

namespace {
constexpr int Bc = 4, Sc = 4096, Ec = 1024, Hc = 16, HDc = 64;
constexpr int Mc = Bc * Sc;          // 16384 tokens
}

// ---------------- WMMA fragment helpers (bf16 16x16x32) ----------------
// A: 16(M) x 32(K). lane<16: row=lane, K = {c..c+7, c+16..c+23}, c=0
//                   lane>=16: row=lane-16, c=8
__device__ __forceinline__ v16bf load_frag_a(const bf16_t* base, int lda, int lane) {
    union { uint4 q[2]; v16bf v; } u;
    int r = lane & 15, h = lane >> 4;
    const bf16_t* p = base + (size_t)r * lda + (h ? 8 : 0);
    u.q[0] = *(const uint4*)(p);
    u.q[1] = *(const uint4*)(p + 16);
    return u.v;
}
// B: 32(K) x 16(N), stored as rows of the [N,K] operand.
// lane<16: col=lane, K={0..7,16..23} ; lane>=16: col=lane-16, K={8..15,24..31}
__device__ __forceinline__ v16bf load_frag_b(const bf16_t* base, int ldb, int lane) {
    union { uint4 q[2]; v16bf v; } u;
    int n = lane & 15, h = lane >> 4;
    const bf16_t* p = base + (size_t)n * ldb + (h ? 8 : 0);
    u.q[0] = *(const uint4*)(p);
    u.q[1] = *(const uint4*)(p + 16);
    return u.v;
}

// One wave computes a 64x64 fp32 tile: C += A[64 x ktot] * B[ktot x 64]
// A row-major [M,K] (lda), B stored as [N,K] rows (ldb).
__device__ __forceinline__ void wave_gemm_64x64(const bf16_t* __restrict__ Abase, int lda,
                                                const bf16_t* __restrict__ Bbase, int ldb,
                                                int ktot, v8f acc[4][4], int lane) {
    for (int k0 = 0; k0 < ktot; k0 += 32) {
        // unconditional speculative prefetch of next K-step of A
        // (global_prefetch_b8; one step past tile end is harmless)
        __builtin_prefetch(Abase + (size_t)(lane & 15) * lda + k0 + 32, 0, 3);
        v16bf af[4], bfv[4];
#pragma unroll
        for (int i = 0; i < 4; i++)
            af[i] = load_frag_a(Abase + (size_t)(i * 16) * lda + k0, lda, lane);
#pragma unroll
        for (int j = 0; j < 4; j++)
            bfv[j] = load_frag_b(Bbase + (size_t)(j * 16) * ldb + k0, ldb, lane);
#pragma unroll
        for (int i = 0; i < 4; i++)
#pragma unroll
            for (int j = 0; j < 4; j++)
                acc[i][j] = __builtin_amdgcn_wmma_f32_16x16x32_bf16(
                    false, af[i], false, bfv[j], (short)0, acc[i][j], false, false);
    }
}

// C layout: lane<16 -> elem r = C[r][lane]; lane>=16 -> elem r = C[8+r][lane-16]
__device__ __forceinline__ void store_tile_bf16(bf16_t* __restrict__ y, int ld,
                                                int m0, int n0, v8f acc[4][4], int lane) {
    int n = lane & 15, hh = lane >> 4;
#pragma unroll
    for (int i = 0; i < 4; i++)
#pragma unroll
        for (int j = 0; j < 4; j++)
#pragma unroll
            for (int r = 0; r < 8; r++) {
                int row = m0 + i * 16 + (hh ? 8 : 0) + r;
                int col = n0 + j * 16 + n;
                y[(size_t)row * ld + col] = (bf16_t)acc[i][j][r];
            }
}

// ---------------- Kernel 1: weight row-norm -> bf16 ----------------
// wn[o,:] = w[o,:] * gain/32 / (eps + ||w[o,:]||/32)    (fan_in = 1024)
__global__ __launch_bounds__(256, 2)
void wnorm_kernel(const float* __restrict__ wq, const float* __restrict__ wk,
                  const float* __restrict__ wv, const float* __restrict__ wo,
                  const float* __restrict__ gain, bf16_t* __restrict__ wn_base) {
    __shared__ float red[256];
    int r = blockIdx.x, widx = blockIdx.y, tid = threadIdx.x;
    const float* w = (widx == 0) ? wq : (widx == 1) ? wk : (widx == 2) ? wv : wo;
    float4 v4 = *(const float4*)(w + (size_t)r * Ec + tid * 4);
    red[tid] = v4.x * v4.x + v4.y * v4.y + v4.z * v4.z + v4.w * v4.w;
    __syncthreads();
    for (int s = 128; s > 0; s >>= 1) {
        if (tid < s) red[tid] += red[tid + s];
        __syncthreads();
    }
    float nrm = sqrtf(red[0]);
    float scale = (*gain) * 0.03125f / (MP_EPS + nrm * 0.03125f);
    bf16_t* out = wn_base + (size_t)widx * Ec * Ec + (size_t)r * Ec + tid * 4;
    out[0] = (bf16_t)(v4.x * scale);
    out[1] = (bf16_t)(v4.y * scale);
    out[2] = (bf16_t)(v4.z * scale);
    out[3] = (bf16_t)(v4.w * scale);
}

// ---------------- Kernel 2: x fp32 -> bf16 ----------------
__global__ __launch_bounds__(256, 4)
void xconv_kernel(const float* __restrict__ x, bf16_t* __restrict__ xb) {
    size_t i = ((size_t)blockIdx.x * 256 + threadIdx.x) * 4;
    float4 v = *(const float4*)(x + i);
    bf16_t o[4] = {(bf16_t)v.x, (bf16_t)v.y, (bf16_t)v.z, (bf16_t)v.w};
    *(uint2*)(xb + i) = *(const uint2*)o;
}

// ---------------- Kernel 3: fused Q/K/V GEMM  y = xb @ wnᵀ ----------------
// grid (M/512, E/64, 3), 256 thr. wave -> 64x64 tile.
__global__ __launch_bounds__(256, 1)
void qkv_gemm_kernel(const bf16_t* __restrict__ xb,
                     const bf16_t* __restrict__ wn_base,
                     bf16_t* __restrict__ y_base) {
    int lane = threadIdx.x & 31, wave = threadIdx.x >> 5;
    int m0 = blockIdx.x * 512 + wave * 64;
    int n0 = blockIdx.y * 64;
    const bf16_t* wsel = wn_base + (size_t)blockIdx.z * Ec * Ec;
    bf16_t* ysel = y_base + (size_t)blockIdx.z * Mc * Ec;
    v8f acc[4][4] = {};
    wave_gemm_64x64(xb + (size_t)m0 * Ec, Ec, wsel + (size_t)n0 * Ec, Ec, Ec, acc, lane);
    store_tile_bf16(ysel, Ec, m0, n0, acc, lane);
}

// ---------------- Kernel 4: q/k head-norm + transpose to [B,H,hd,S] ----------------
// grid (S/64, H, B*2): z = b*2 + isK. k additionally scaled by 1/sqrt(S)=1/64.
__global__ __launch_bounds__(256, 2)
void qk_norm_transpose_kernel(const bf16_t* __restrict__ y_base,
                              bf16_t* __restrict__ qT, bf16_t* __restrict__ kT) {
    __shared__ float t[64][65];
    __shared__ float rscale[64];
    int tid = threadIdx.x;
    int s0 = blockIdx.x * 64, h = blockIdx.y;
    int b = blockIdx.z >> 1, isK = blockIdx.z & 1;
    const bf16_t* y = y_base + (isK ? (size_t)Mc * Ec : 0);
    bf16_t* outT = (isK ? kT : qT) + (size_t)((b * Hc + h) * HDc) * Sc;
#pragma unroll
    for (int it = 0; it < 16; it++) {
        int idx = tid + it * 256;
        int i = idx >> 6, j = idx & 63;
        t[i][j] = (float)y[(size_t)(b * Sc + s0 + i) * Ec + h * HDc + j];
    }
    __syncthreads();
    if (tid < 64) {
        float ss = 0.f;
#pragma unroll 8
        for (int j = 0; j < 64; j++) { float v = t[tid][j]; ss += v * v; }
        float sc = 1.0f / (MP_EPS + sqrtf(ss) * 0.125f);   // /sqrt(64)
        if (isK) sc *= (1.0f / 64.0f);                     // 1/sqrt(S)
        rscale[tid] = sc;
    }
    __syncthreads();
#pragma unroll
    for (int it = 0; it < 16; it++) {
        int idx = tid + it * 256;
        int d = idx >> 6, i = idx & 63;
        outT[(size_t)d * Sc + s0 + i] = (bf16_t)(t[i][d] * rscale[i]);
    }
}

// ---------------- Kernel 5: scores = q̂ᵀ k̂' (contract S) + softmax ----------------
// grid (B*H), 256 thr. 8 waves split K=4096, reduce via ds_add_f32.
__global__ __launch_bounds__(256, 1)
void scores_softmax_kernel(const bf16_t* __restrict__ qT,
                           const bf16_t* __restrict__ kT,
                           bf16_t* __restrict__ attn) {
    __shared__ float sc[64][64];
    int tid = threadIdx.x, lane = tid & 31, wave = tid >> 5;
    int bh = blockIdx.x;
    const bf16_t* Aq = qT + (size_t)bh * HDc * Sc;
    const bf16_t* Bk = kT + (size_t)bh * HDc * Sc;
    for (int i = tid; i < 64 * 64; i += 256) ((float*)sc)[i] = 0.f;
    __syncthreads();
    v8f acc[4][4] = {};
    int kchunk = wave * 512;
    wave_gemm_64x64(Aq + kchunk, Sc, Bk + kchunk, Sc, 512, acc, lane);
    int n = lane & 15, hh = lane >> 4;
#pragma unroll
    for (int i = 0; i < 4; i++)
#pragma unroll
        for (int j = 0; j < 4; j++)
#pragma unroll
            for (int r = 0; r < 8; r++)
                atomicAdd(&sc[i * 16 + (hh ? 8 : 0) + r][j * 16 + n], acc[i][j][r]);
    __syncthreads();
    if (tid < 64) {
        float mx = -1e30f;
        for (int j = 0; j < 64; j++) mx = fmaxf(mx, sc[tid][j]);
        float sum = 0.f;
        for (int j = 0; j < 64; j++) { float e = __expf(sc[tid][j] - mx); sc[tid][j] = e; sum += e; }
        float inv = 1.0f / sum;
        bf16_t* out = attn + (size_t)bh * 64 * 64 + tid * 64;
        for (int j = 0; j < 64; j++) out[j] = (bf16_t)(sc[tid][j] * inv);
    }
}

// ---------------- Kernel 6: o = v @ attnᵀ, written as [B,S,E] bf16 ----------------
// grid (S/512, H, B)
__global__ __launch_bounds__(256, 1)
void o_gemm_kernel(const bf16_t* __restrict__ yv, const bf16_t* __restrict__ attn,
                   bf16_t* __restrict__ ob) {
    int lane = threadIdx.x & 31, wave = threadIdx.x >> 5;
    int b = blockIdx.z, h = blockIdx.y;
    int s0 = blockIdx.x * 512 + wave * 64;
    const bf16_t* A  = yv + (size_t)(b * Sc + s0) * Ec + h * HDc;       // [64 x 64], lda=E
    const bf16_t* Bp = attn + (size_t)(b * Hc + h) * HDc * HDc;        // attn[h_out, k] rows
    v8f acc[4][4] = {};
    wave_gemm_64x64(A, Ec, Bp, HDc, HDc, acc, lane);
    store_tile_bf16(ob, Ec, b * Sc + s0, h * HDc, acc, lane);
}

// ---------------- Kernel 7: out = (res + o @ wn_oᵀ) / sqrt(2), fp32 ----------------
// grid (M/512, E/64)
__global__ __launch_bounds__(256, 1)
void out_gemm_kernel(const bf16_t* __restrict__ ob, const bf16_t* __restrict__ wno,
                     const float* __restrict__ query, float* __restrict__ out) {
    int lane = threadIdx.x & 31, wave = threadIdx.x >> 5;
    int m0 = blockIdx.x * 512 + wave * 64;
    int n0 = blockIdx.y * 64;
    v8f acc[4][4] = {};
    wave_gemm_64x64(ob + (size_t)m0 * Ec, Ec, wno + (size_t)n0 * Ec, Ec, Ec, acc, lane);
    int n = lane & 15, hh = lane >> 4;
#pragma unroll
    for (int i = 0; i < 4; i++)
#pragma unroll
        for (int j = 0; j < 4; j++)
#pragma unroll
            for (int r = 0; r < 8; r++) {
                int row = m0 + i * 16 + (hh ? 8 : 0) + r;
                int col = n0 + j * 16 + n;
                size_t idx = (size_t)row * Ec + col;
                out[idx] = (query[idx] + acc[i][j][r]) * 0.70710678118654752f;
            }
}

extern "C" void kernel_launch(void* const* d_in, const int* in_sizes, int n_in,
                              void* d_out, int out_size, void* d_ws, size_t ws_size,
                              hipStream_t stream) {
    const float* x    = (const float*)d_in[0];
    const float* gain = (const float*)d_in[1];
    const float* wq   = (const float*)d_in[2];
    const float* wk   = (const float*)d_in[3];
    const float* wv   = (const float*)d_in[4];
    const float* wo   = (const float*)d_in[5];
    float* out = (float*)d_out;

    char* ws = (char*)d_ws;
    size_t off = 0;
    bf16_t* xb = (bf16_t*)(ws + off);  off += (size_t)Mc * Ec * 2;          // 32 MB
    bf16_t* wn = (bf16_t*)(ws + off);  off += (size_t)4 * Ec * Ec * 2;      //  8 MB (q,k,v,o contiguous)
    bf16_t* yq = (bf16_t*)(ws + off);  off += (size_t)3 * Mc * Ec * 2;      // 96 MB (yq,yk,yv contiguous)
    bf16_t* qT = (bf16_t*)(ws + off);  off += (size_t)Bc * Hc * HDc * Sc * 2; // 32 MB
    bf16_t* kT = (bf16_t*)(ws + off);  off += (size_t)Bc * Hc * HDc * Sc * 2; // 32 MB
    bf16_t* attn = (bf16_t*)(ws + off); off += (size_t)Bc * Hc * HDc * HDc * 2;
    bf16_t* yv = yq + (size_t)2 * Mc * Ec;
    bf16_t* ob = yq;   // reuse yq: consumed by relayout before o_gemm writes it

    wnorm_kernel<<<dim3(Ec, 4), 256, 0, stream>>>(wq, wk, wv, wo, gain, wn);
    xconv_kernel<<<dim3((Mc * Ec) / 1024), 256, 0, stream>>>(x, xb);
    qkv_gemm_kernel<<<dim3(Mc / 512, Ec / 64, 3), 256, 0, stream>>>(xb, wn, yq);
    qk_norm_transpose_kernel<<<dim3(Sc / 64, Hc, Bc * 2), 256, 0, stream>>>(yq, qT, kT);
    scores_softmax_kernel<<<dim3(Bc * Hc), 256, 0, stream>>>(qT, kT, attn);
    o_gemm_kernel<<<dim3(Sc / 512, Hc, Bc), 256, 0, stream>>>(yv, attn, ob);
    out_gemm_kernel<<<dim3(Mc / 512, Ec / 64), 256, 0, stream>>>(ob, wn + (size_t)3 * Ec * Ec, x, out);
}
